// Head_35553739276740
// MI455X (gfx1250) — matrix operand order repaired and checked by reading, not compiled, not run on hardware
//
#include <hip/hip_runtime.h>

#define B_ 4
#define T_ 4096
#define C_ 1024
#define H_ 64
#define SCL2 0.015625f   /* (HEAD_SIZE^-0.5)^2 = 1/64 : double scaling, faithful */
#define NEGINF -3.0e30f

typedef _Float16 half_t;
typedef half_t v16h __attribute__((ext_vector_type(16)));
typedef half_t v8h  __attribute__((ext_vector_type(8)));
typedef half_t v4h  __attribute__((ext_vector_type(4)));
typedef float  v8f  __attribute__((ext_vector_type(8)));
typedef float  v4f  __attribute__((ext_vector_type(4)));
typedef unsigned int u32x4 __attribute__((ext_vector_type(4)));
typedef int i32x4 __attribute__((ext_vector_type(4)));
typedef int i32x8 __attribute__((ext_vector_type(8)));

#if defined(__has_builtin)
#if __has_builtin(__builtin_amdgcn_tensor_load_to_lds)
#define HAVE_TDM 1
#endif
#endif

static __device__ __forceinline__ v16h cat8(v8h lo, v8h hi) {
  return __builtin_shufflevector(lo, hi, 0,1,2,3,4,5,6,7,8,9,10,11,12,13,14,15);
}

// f16 A-operand chunk from 16 f32: e=0..7 -> p[0..7], e=8..15 -> p2[0..7]
static __device__ __forceinline__ v16h load_a16_f32(const float* p, const float* p2) {
  v4f a0 = *(const v4f*)(p);
  v4f a1 = *(const v4f*)(p + 4);
  v4f a2 = *(const v4f*)(p2);
  v4f a3 = *(const v4f*)(p2 + 4);
  v16h r;
#pragma unroll
  for (int e = 0; e < 4; ++e) {
    r[e]      = (half_t)a0[e];
    r[4 + e]  = (half_t)a1[e];
    r[8 + e]  = (half_t)a2[e];
    r[12 + e] = (half_t)a3[e];
  }
  return r;
}

#define WMMA_F16(A, Bm, Cm) \
  __builtin_amdgcn_wmma_f32_16x16x32_f16(false, (A), false, (Bm), (short)0, (Cm), false, false)

#ifdef HAVE_TDM
// Issue a TDM 2-D tile load (f16 elements) : global -> LDS.
// Descriptor layout per CDNA5 ISA ch.8 (groups 0/1; groups 2/3 zero for 2-D).
static __device__ __forceinline__ void tdm_load_2d(
    unsigned lds_addr, const half_t* gptr,
    int tensor_d0, int tensor_d1, int tile_d0, int tile_d1, int stride0)
{
  unsigned long long ga = (unsigned long long)(size_t)gptr;
  u32x4 g0;
  g0[0] = 1u;                                   // count=1, user descriptor
  g0[1] = lds_addr;                             // lds_addr [63:32]
  g0[2] = (unsigned)ga;                         // global_addr[31:0]
  g0[3] = (unsigned)((ga >> 32) & 0x1FFFFFFu) | (2u << 30);   // addr[56:32] | type=2
  i32x8 g1;
  g1[0] = 0x00010000;                           // data_size=1 (2 bytes), no pad/iter
  g1[1] = (tensor_d0 & 0xffff) << 16;           // tensor_dim0[15:0] @ [63:48]
  g1[2] = ((tensor_d0 >> 16) & 0xffff) | ((tensor_d1 & 0xffff) << 16);
  g1[3] = ((tensor_d1 >> 16) & 0xffff) | ((tile_d0 & 0xffff) << 16);
  g1[4] = (tile_d1 & 0xffff);                   // tile_dim1 ; tile_dim2 = 0
  g1[5] = stride0;                              // tensor_dim0_stride[31:0]
  g1[6] = 0;                                    // stride0 hi / stride1 lo
  g1[7] = 0;
  i32x4 gz = {0, 0, 0, 0};
#if __clang_major__ <= 22
  __builtin_amdgcn_tensor_load_to_lds(g0, g1, gz, gz, 0);
#else
  i32x8 gz8 = {0, 0, 0, 0, 0, 0, 0, 0};
  __builtin_amdgcn_tensor_load_to_lds(g0, g1, gz, gz, gz8, 0);
#endif
}
#endif

// ---------------------------------------------------------------------------
// Phase 0: convert weights f32 -> f16 once.  W16 layout: [3][H][C]
// ---------------------------------------------------------------------------
__global__ __launch_bounds__(256) void wcvt_kernel(
    const float* __restrict__ Wk, const float* __restrict__ Wq,
    const float* __restrict__ Wv, half_t* __restrict__ W16)
{
  const int idx = blockIdx.x * 256 + threadIdx.x;     // 49152 threads * 4 elems
  const int w = idx / (H_ * C_ / 4);
  const int r = idx % (H_ * C_ / 4);
  const float* s = (w == 0) ? Wk : ((w == 1) ? Wq : Wv);
  v4f f = *(const v4f*)(s + (size_t)r * 4);
  v4h h;
#pragma unroll
  for (int e = 0; e < 4; ++e) h[e] = (half_t)f[e];
  *(v4h*)(W16 + (size_t)w * H_ * C_ + (size_t)r * 4) = h;
}

// ---------------------------------------------------------------------------
// Phase 1: K = x@Wk^T, Q = x@Wq^T, V = x@Wv^T.
// One wave per 16-row tile; K/Q stored [B,T,64], V stored transposed [B,64,T].
// Inner loop is software pipelined: all 12 B-operands batched ahead of the
// 12-WMMA chain; next k-step's A staged during the chain.
// ---------------------------------------------------------------------------
__global__ __launch_bounds__(32) void proj_kernel(
    const float* __restrict__ x,     // [B,T,C]
    const half_t* __restrict__ W16,  // [3,H,C] f16
    half_t* __restrict__ Kf,         // [B,T,H]
    half_t* __restrict__ Qf,         // [B,T,H]
    half_t* __restrict__ Vt)         // [B,H,T]
{
  const int lane = threadIdx.x;
  const int g  = lane >> 4;
  const int ln = lane & 15;
  const int nt = T_ / 16;
  const int b  = blockIdx.x / nt;
  const int i0 = (blockIdx.x % nt) * 16;

  const v8f vzero = {0.f,0.f,0.f,0.f,0.f,0.f,0.f,0.f};
  v8f acc[12];
#pragma unroll
  for (int wt = 0; wt < 12; ++wt) acc[wt] = vzero;

  const float*  xrow  = x + ((size_t)b * T_ + i0 + ln) * C_;
  const half_t* wbase = W16 + (size_t)ln * C_ + 16 * g;

  v16h A = load_a16_f32(xrow + 16 * g, xrow + 8 + 16 * g);

  for (int kk = 0; kk < C_ / 32; ++kk) {
    const int cb = kk * 32;
    // batch all 12 B operands (contiguous 32-byte f16 loads) for this k-step
    v16h Bop[12];
#pragma unroll
    for (int w = 0; w < 3; ++w)
#pragma unroll
      for (int t = 0; t < 4; ++t)
        Bop[w * 4 + t] = *(const v16h*)(wbase + (size_t)w * H_ * C_ +
                                        (size_t)(16 * t) * C_ + cb);
    // stage next A while the WMMA chain drains
    v16h An = A;
    if (kk + 1 < C_ / 32)
      An = load_a16_f32(xrow + cb + 32 + 16 * g, xrow + cb + 40 + 16 * g);
#pragma unroll
    for (int wt = 0; wt < 12; ++wt)
      acc[wt] = WMMA_F16(A, Bop[wt], acc[wt]);
    A = An;
  }

  const size_t rowbase = (size_t)b * T_ + i0;
#pragma unroll
  for (int t = 0; t < 4; ++t) {
#pragma unroll
    for (int r = 0; r < 8; ++r) {
      const int row = r + 8 * g;
      Kf[(rowbase + row) * H_ + 16 * t + ln] = (half_t)acc[0 + t][r];
      Qf[(rowbase + row) * H_ + 16 * t + ln] = (half_t)acc[4 + t][r];
      Vt[((size_t)b * H_ + 16 * t + ln) * T_ + i0 + row] = (half_t)acc[8 + t][r];
    }
  }
}

// ---------------------------------------------------------------------------
// Phase 2: flash attention with online softmax.
// One wave per (batch, 16-query-row tile). 32 keys per iteration; K/V tiles
// staged into LDS by the Tensor Data Mover (double buffered, TENSORcnt sync).
// ---------------------------------------------------------------------------
__global__ __launch_bounds__(32) void attn_kernel(
    const half_t* __restrict__ Qf,  // [B,T,H]
    const half_t* __restrict__ Kf,  // [B,T,H]
    const half_t* __restrict__ Vt,  // [B,H,T]
    float* __restrict__ out)        // [B,T,H]
{
  __shared__ half_t P[16 * 32];          // S->P transpose staging
  __shared__ half_t Kbuf[2][32 * 64];    // [key][dim]
  __shared__ half_t Vbuf[2][64 * 32];    // [dim][key]

  const int lane = threadIdx.x;
  const int g  = lane >> 4;
  const int ln = lane & 15;
  const int nt = T_ / 16;
  const int b  = blockIdx.x / nt;
  const int i0 = (blockIdx.x % nt) * 16;

  // Q tile as A-operands for both 32-wide K-steps of the 64-dim head
  const half_t* qrow = Qf + ((size_t)b * T_ + i0 + ln) * H_;
  const v16h a0 = cat8(*(const v8h*)(qrow + 16 * g),      *(const v8h*)(qrow + 8 + 16 * g));
  const v16h a1 = cat8(*(const v8h*)(qrow + 32 + 16 * g), *(const v8h*)(qrow + 40 + 16 * g));

  const v8f vzero = {0.f,0.f,0.f,0.f,0.f,0.f,0.f,0.f};
  v8f o[4];
#pragma unroll
  for (int t = 0; t < 4; ++t) o[t] = vzero;
  float rmax[8], rsum[8];
#pragma unroll
  for (int r = 0; r < 8; ++r) { rmax[r] = NEGINF; rsum[r] = 0.f; }

  int jmax = i0 + 16;
  if (jmax > T_ - 1) jmax = T_ - 1;
  const int nblk = (jmax >> 5) + 1;      // 32-key blocks

#ifdef HAVE_TDM
  // prologue: stage block 0 into buffer 0
  tdm_load_2d((unsigned)(size_t)&Kbuf[0][0], Kf + ((size_t)b * T_) * H_,
              H_, T_, H_, 32, H_);
  tdm_load_2d((unsigned)(size_t)&Vbuf[0][0], Vt + (size_t)b * H_ * T_,
              T_, H_, 32, H_, T_);
#endif

  for (int jb = 0; jb < nblk; ++jb) {
    const int j0 = jb * 32;
    const int cur = jb & 1;

#ifdef HAVE_TDM
    if (jb + 1 < nblk) {   // stage next block into the other buffer
      const int j1 = j0 + 32;
      tdm_load_2d((unsigned)(size_t)&Kbuf[cur ^ 1][0],
                  Kf + ((size_t)b * T_ + j1) * H_, H_, T_, H_, 32, H_);
      tdm_load_2d((unsigned)(size_t)&Vbuf[cur ^ 1][0],
                  Vt + (size_t)b * H_ * T_ + j1, T_, H_, 32, H_, T_);
      __builtin_amdgcn_s_wait_tensorcnt(2);   // current block's 2 TDM ops done
    } else {
      __builtin_amdgcn_s_wait_tensorcnt(0);
    }
    const half_t* Kl = &Kbuf[cur][0];
    const half_t* Vl = &Vbuf[cur][0];
#endif

    // ---- S = Q K^T : two 16x16 tiles --------------------------------------
    v8f s[2];
#pragma unroll
    for (int t = 0; t < 2; ++t) {
#ifdef HAVE_TDM
      const half_t* krow = Kl + (16 * t + ln) * H_ + 16 * g;
#else
      const half_t* krow = Kf + ((size_t)b * T_ + j0 + 16 * t + ln) * H_ + 16 * g;
#endif
      v16h bk0 = *(const v16h*)(krow);       // head dims 16g..16g+15
      v16h bk1 = *(const v16h*)(krow + 32);  // head dims 32+16g..
      v8f sa = WMMA_F16(a0, bk0, vzero);
      sa = WMMA_F16(a1, bk1, sa);
      s[t] = sa;
    }

    // ---- masked logits + online softmax -----------------------------------
    float p0[8], p1[8];
#pragma unroll
    for (int r = 0; r < 8; ++r) {
      const int row = i0 + r + 8 * g;
      float s0v = s[0][r] * SCL2; if (j0 + ln      > row + 1) s0v = NEGINF;
      float s1v = s[1][r] * SCL2; if (j0 + 16 + ln > row + 1) s1v = NEGINF;
      float m = fmaxf(s0v, s1v);
      m = fmaxf(m, __shfl_xor(m, 1, 32));
      m = fmaxf(m, __shfl_xor(m, 2, 32));
      m = fmaxf(m, __shfl_xor(m, 4, 32));
      m = fmaxf(m, __shfl_xor(m, 8, 32));
      const float nm = fmaxf(rmax[r], m);
      const float sf = __expf(rmax[r] - nm);
      rmax[r] = nm;
      const float e0 = __expf(s0v - nm);
      const float e1 = __expf(s1v - nm);
      p0[r] = e0; p1[r] = e1;
      float ps = e0 + e1;
      ps += __shfl_xor(ps, 1, 32);
      ps += __shfl_xor(ps, 2, 32);
      ps += __shfl_xor(ps, 4, 32);
      ps += __shfl_xor(ps, 8, 32);
      rsum[r] = rsum[r] * sf + ps;
#pragma unroll
      for (int t = 0; t < 4; ++t) o[t][r] *= sf;
    }

    // ---- D-layout -> A-layout transpose of P through LDS ------------------
#pragma unroll
    for (int r = 0; r < 8; ++r) {
      P[(r + 8 * g) * 32 + ln]      = (half_t)p0[r];
      P[(r + 8 * g) * 32 + 16 + ln] = (half_t)p1[r];
    }
    __syncthreads();
    const v16h pa = cat8(*(const v8h*)(&P[ln * 32 + 16 * g]),
                         *(const v8h*)(&P[ln * 32 + 8 + 16 * g]));
    __syncthreads();

    // ---- O += P V ---------------------------------------------------------
#pragma unroll
    for (int t = 0; t < 4; ++t) {
#ifdef HAVE_TDM
      const half_t* vp = Vl + (16 * t + ln) * 32 + 16 * g;
#else
      const half_t* vp = Vt + ((size_t)b * H_ + 16 * t + ln) * T_ + j0 + 16 * g;
#endif
      v16h bv = *(const v16h*)vp;   // keys (tile-local) 16g..16g+15
      o[t] = WMMA_F16(pa, bv, o[t]);
    }
  }

  // ---- normalize + write --------------------------------------------------
  const size_t ob = ((size_t)b * T_ + i0) * H_;
#pragma unroll
  for (int r = 0; r < 8; ++r) {
    const float inv = 1.0f / rsum[r];
#pragma unroll
    for (int t = 0; t < 4; ++t)
      out[ob + (size_t)(r + 8 * g) * H_ + 16 * t + ln] = o[t][r] * inv;
  }
}

// ---------------------------------------------------------------------------
extern "C" void kernel_launch(void* const* d_in, const int* in_sizes, int n_in,
                              void* d_out, int out_size, void* d_ws, size_t ws_size,
                              hipStream_t stream) {
  (void)in_sizes; (void)n_in; (void)out_size; (void)ws_size;
  const float* x  = (const float*)d_in[0];
  const float* Wk = (const float*)d_in[1];
  const float* Wq = (const float*)d_in[2];
  const float* Wv = (const float*)d_in[3];

  const size_t nkv = (size_t)B_ * T_ * H_;   // 1 Mi halves each
  half_t* Kf  = (half_t*)d_ws;
  half_t* Qf  = Kf + nkv;
  half_t* Vt  = Qf + nkv;
  half_t* W16 = Vt + nkv;                    // 3*H*C halves (384 KB) ; total < 6.5 MB

  wcvt_kernel<<<(3 * H_ * C_ / 4 + 255) / 256, 256, 0, stream>>>(Wk, Wq, Wv, W16);

  const int ntiles = B_ * (T_ / 16);         // 1024 workgroups, 1 wave each
  proj_kernel<<<ntiles, 32, 0, stream>>>(x, W16, Kf, Qf, Vt);
  attn_kernel<<<ntiles, 32, 0, stream>>>(Qf, Kf, Vt, (float*)d_out);
}
